// RouteWrap_72275709657448
// MI455X (gfx1250) — matrix operand order repaired and checked by reading, not compiled, not run on hardware
//
#include <hip/hip_runtime.h>
#include <hip/hip_bf16.h>

// ---------------------------------------------------------------------------
// Problem constants (from reference setup_inputs):
//   b=4, s=2048 -> T = 8192 tokens; h=2048; j=2048; a=64; r=16; TOPK=4
// ---------------------------------------------------------------------------
#define T_TOK   8192
#define H_DIM   2048
#define J_DIM   2048
#define N_ADPT  64
#define R_RANK  16
#define AR_DIM  (N_ADPT * R_RANK)   // 1024
#define TOPK    4

typedef __attribute__((ext_vector_type(16))) __bf16        v16bf;
typedef __attribute__((ext_vector_type(8)))  float         v8f;
typedef __attribute__((ext_vector_type(4)))  unsigned int  u32x4;

// ---------------------------------------------------------------------------
// fp32 -> bf16 elementwise conversion (row-major preserved) for x
// ---------------------------------------------------------------------------
__global__ void cvt_f32_bf16_kernel(const float* __restrict__ in,
                                    __bf16* __restrict__ out, int n) {
    int i = blockIdx.x * blockDim.x + threadIdx.x;
    if (i < n) out[i] = (__bf16)in[i];
}

// ---------------------------------------------------------------------------
// fp32 [Nr][K] -> bf16 [K][Nr] tiled transpose-convert (coalesced both ways).
// Block (32,8); grid (K/32, Nr/32).
// ---------------------------------------------------------------------------
__global__ __launch_bounds__(256)
void cvt_transpose_kernel(const float* __restrict__ in,
                          __bf16* __restrict__ out, int Nr, int K) {
    __shared__ __bf16 tile[32][33];
    const int kb = blockIdx.x * 32;
    const int nb = blockIdx.y * 32;
#pragma unroll
    for (int i = 0; i < 4; ++i) {
        const int nr = nb + threadIdx.y + i * 8;
        tile[threadIdx.y + i * 8][threadIdx.x] =
            (__bf16)in[(size_t)nr * K + kb + threadIdx.x];
    }
    __syncthreads();
#pragma unroll
    for (int i = 0; i < 4; ++i) {
        const int kr = kb + threadIdx.y + i * 8;
        out[(size_t)kr * Nr + nb + threadIdx.x] = tile[threadIdx.x][threadIdx.y + i * 8];
    }
}

// ---------------------------------------------------------------------------
// GEMM with bf16 WMMA:  C[M,N] (f32) = X[M,K] (bf16, row-major)
//                                    * Yt[K,N] (bf16, K-major / pre-transposed)
//                                    + bias
// Block = 256 threads (8 waves). Each wave computes a 32x64 tile of C:
// 2 A-fragments x 4 B-fragments -> 8 accumulators, so 4 global_load_b128 (A) + 64 16-bit loads (B)
// feed 8 v_wmma per K-step (0.75 KB of operand traffic per matrix op).
// Waves arranged 4(M) x 2(N) -> block covers 128(M) x 128(N).
// No LDS, no barriers in the hot loop; cross-block reuse rides the 192 MB L2.
//
// Fragment layouts (CDNA5 ISA 7.12.2, wave32, 16x16x32 bf16):
//  A (16x32, MxK): lane L holds row M=L%16; lanes 0-15 hold
//     K = {0..7, 16..23}, lanes 16-31 hold K = {8..15, 24..31}
//     -> two 16B contiguous loads from the row-major X.
//  B (32x16, KxN): lane L holds column N=L%16; K split as for A
//     -> 16 strided 2-byte loads (stride N) from K-major Yt.
//  C/D (16x16 f32): VGPR i -> M=i (lanes 0-15) / M=i+8 (lanes 16-31),
//     N = lane%16.
// ---------------------------------------------------------------------------
__global__ __launch_bounds__(256)
void gemm_nt_bf16_wmma_kernel(const __bf16* __restrict__ X,
                              const __bf16* __restrict__ Yt,
                              const float* __restrict__ bias,
                              float* __restrict__ C,
                              int M, int N, int K) {
    const int lane  = threadIdx.x & 31;
    const int wid   = threadIdx.x >> 5;
    const int Mtile = blockIdx.y * 128 + (wid & 3) * 32;
    const int Ntile = blockIdx.x * 128 + (wid >> 2) * 64;

    union Frag { v16bf v; u32x4 q[2]; };

    v8f acc[2][4];
#pragma unroll
    for (int a = 0; a < 2; ++a)
#pragma unroll
        for (int b = 0; b < 4; ++b)
            acc[a][b] = (v8f){};

    const int m0   = Mtile + (lane & 15);
    const int kofs = (lane < 16) ? 0 : 8;

    const __bf16* xptr0 = X + (size_t)m0 * K + kofs;          // A frag, rows 0..15
    const __bf16* xptr1 = xptr0 + (size_t)16 * K;             // A frag, rows 16..31
    const __bf16* yptr  = Yt + (size_t)kofs * N + Ntile + (lane & 15);  // B frags (n=lane%16, k table order)

#pragma unroll 2
    for (int kk = 0; kk < K; kk += 32) {
        Frag fa0, fa1, fb[4];
        fa0.q[0] = *(const u32x4*)(xptr0);
        fa0.q[1] = *(const u32x4*)(xptr0 + 16);
        fa1.q[0] = *(const u32x4*)(xptr1);
        fa1.q[1] = *(const u32x4*)(xptr1 + 16);
#pragma unroll
        for (int b = 0; b < 4; ++b) {
#pragma unroll
            for (int s = 0; s < 8; ++s) {
                fb[b].v[s]     = yptr[(size_t)s * N + b * 16];
                fb[b].v[s + 8] = yptr[(size_t)(s + 16) * N + b * 16];
            }
        }

#pragma unroll
        for (int b = 0; b < 4; ++b) {
            acc[0][b] = __builtin_amdgcn_wmma_f32_16x16x32_bf16(
                false, fa0.v, false, fb[b].v, (short)0, acc[0][b], false, false);
            acc[1][b] = __builtin_amdgcn_wmma_f32_16x16x32_bf16(
                false, fa1.v, false, fb[b].v, (short)0, acc[1][b], false, false);
        }

        xptr0 += 32;
        xptr1 += 32;
        yptr  += (size_t)32 * N;
    }

    // --- writeout per C/D layout ---
    const int mbase = Mtile + ((lane < 16) ? 0 : 8);
#pragma unroll
    for (int b = 0; b < 4; ++b) {
        const int ncol = Ntile + b * 16 + (lane & 15);
        const float bv = bias ? bias[ncol] : 0.0f;
#pragma unroll
        for (int a = 0; a < 2; ++a) {
            const int mrow = mbase + a * 16;
#pragma unroll
            for (int i = 0; i < 8; ++i)
                C[(size_t)(mrow + i) * N + ncol] = acc[a][b][i] + bv;
        }
    }
}

// ---------------------------------------------------------------------------
// Routing: per token, squared L2 norms over r (monotone with norm -> same
// top-k), 4-pass argmax (ties -> lowest index, matching jax.lax.top_k),
// emit v[t,r] = 0.25 * sum of selected vecs, and sel[t,0..3].
// One 64-thread block per token.
// ---------------------------------------------------------------------------
__global__ __launch_bounds__(64)
void route_kernel(const float* __restrict__ vecs,   // [T, AR_DIM]
                  float* __restrict__ v_out,        // [T, R_RANK]
                  int* __restrict__ sel_out) {      // [T, TOPK]
    const int t   = blockIdx.x;
    const int tid = threadIdx.x;

    __shared__ float sq[N_ADPT];
    __shared__ int   sel[TOPK];

    const float* vt = vecs + (size_t)t * AR_DIM;

    float s = 0.0f;
#pragma unroll
    for (int r = 0; r < R_RANK; ++r) {
        float x = vt[tid * R_RANK + r];
        s += x * x;
    }
    sq[tid] = s;
    __syncthreads();

    if (tid == 0) {
#pragma unroll
        for (int i = 0; i < TOPK; ++i) {
            float best = -1.0f; int bi = 0;
            for (int a = 0; a < N_ADPT; ++a)
                if (sq[a] > best) { best = sq[a]; bi = a; }
            sel[i] = bi;
            sq[bi] = -1.0f;
            sel_out[t * TOPK + i] = bi;
        }
    }
    __syncthreads();

    if (tid < R_RANK) {
        float acc = 0.0f;
#pragma unroll
        for (int i = 0; i < TOPK; ++i)
            acc += vt[sel[i] * R_RANK + tid];
        v_out[t * R_RANK + tid] = 0.25f * acc;
    }
}

// ---------------------------------------------------------------------------
// delta[t,j] = 0.25 * sum_{i<4} dot(B[sel_i, j, :], v[t,:]); accumulate into
// out (which already holds base + bias). B is 8 MB -> L2 resident.
// Grid: (J/256, T), block 256.
// ---------------------------------------------------------------------------
__global__ __launch_bounds__(256)
void delta_kernel(const float* __restrict__ Bk,     // [N_ADPT, J_DIM, R_RANK]
                  const float* __restrict__ v_in,   // [T, R_RANK]
                  const int* __restrict__ sel_in,   // [T, TOPK]
                  float* __restrict__ out) {        // [T, J_DIM]
    const int t = blockIdx.y;
    const int j = blockIdx.x * blockDim.x + threadIdx.x;

    __shared__ float vs[R_RANK];
    __shared__ int   ss[TOPK];
    if (threadIdx.x < R_RANK) vs[threadIdx.x] = v_in[t * R_RANK + threadIdx.x];
    if (threadIdx.x < TOPK)   ss[threadIdx.x] = sel_in[t * TOPK + threadIdx.x];
    __syncthreads();

    float acc = 0.0f;
#pragma unroll
    for (int i = 0; i < TOPK; ++i) {
        const float* br = Bk + ((size_t)ss[i] * J_DIM + j) * R_RANK;
#pragma unroll
        for (int r = 0; r < R_RANK; ++r)
            acc += br[r] * vs[r];
    }
    out[(size_t)t * J_DIM + j] += 0.25f * acc;
}

// ---------------------------------------------------------------------------
// Launch
// ---------------------------------------------------------------------------
extern "C" void kernel_launch(void* const* d_in, const int* in_sizes, int n_in,
                              void* d_out, int out_size, void* d_ws, size_t ws_size,
                              hipStream_t stream) {
    const float* x    = (const float*)d_in[0];   // [T, H]
    const float* A    = (const float*)d_in[1];   // [a, r, H] == [AR, H] row-major
    const float* Bk   = (const float*)d_in[2];   // [a, J, r]
    const float* W    = (const float*)d_in[3];   // [J, H]
    const float* bias = (const float*)d_in[4];   // [J]
    float* out        = (float*)d_out;           // [T, J]

    // workspace carving (256B aligned)
    char* ws = (char*)d_ws;
    size_t off = 0;
    auto carve = [&](size_t bytes) {
        void* p = ws + off;
        off += (bytes + 255) & ~(size_t)255;
        return p;
    };
    __bf16* xb   = (__bf16*)carve((size_t)T_TOK * H_DIM * sizeof(__bf16));   // 32 MB
    __bf16* Abt  = (__bf16*)carve((size_t)H_DIM * AR_DIM * sizeof(__bf16));  //  4 MB  [K][AR]
    __bf16* Wbt  = (__bf16*)carve((size_t)H_DIM * J_DIM * sizeof(__bf16));   //  8 MB  [K][J]
    float*  vecs = (float*) carve((size_t)T_TOK * AR_DIM * sizeof(float));   // 32 MB
    float*  vbuf = (float*) carve((size_t)T_TOK * R_RANK * sizeof(float));   // 512 KB
    int*    sel  = (int*)   carve((size_t)T_TOK * TOPK * sizeof(int));       // 128 KB

    // 1) bf16 conversions (x row-major; A, W transposed to K-major)
    {
        const int nx = T_TOK * H_DIM;
        cvt_f32_bf16_kernel<<<(nx + 255) / 256, 256, 0, stream>>>(x, xb, nx);

        dim3 blk(32, 8);
        dim3 ga(H_DIM / 32, AR_DIM / 32);
        cvt_transpose_kernel<<<ga, blk, 0, stream>>>(A, Abt, AR_DIM, H_DIM);
        dim3 gw(H_DIM / 32, J_DIM / 32);
        cvt_transpose_kernel<<<gw, blk, 0, stream>>>(W, Wbt, J_DIM, H_DIM);
    }

    // 2) vecs = x . A^T   : [T,H] x [H,AR] -> [T,AR]
    {
        dim3 grid(AR_DIM / 128, T_TOK / 128);
        gemm_nt_bf16_wmma_kernel<<<grid, 256, 0, stream>>>(
            xb, Abt, nullptr, vecs, T_TOK, AR_DIM, H_DIM);
    }

    // 3) routing (norms -> top-4 -> v, sel)
    route_kernel<<<T_TOK, 64, 0, stream>>>(vecs, vbuf, sel);

    // 4) base = x . W^T + bias : [T,H] x [H,J] -> out [T,J]
    {
        dim3 grid(J_DIM / 128, T_TOK / 128);
        gemm_nt_bf16_wmma_kernel<<<grid, 256, 0, stream>>>(
            xb, Wbt, bias, out, T_TOK, J_DIM, H_DIM);
    }

    // 5) out += delta (routed low-rank up-projection)
    {
        dim3 grid(J_DIM / 256, T_TOK);
        delta_kernel<<<grid, 256, 0, stream>>>(Bk, vbuf, sel, out);
    }
}
